// Net_48000554500167
// MI455X (gfx1250) — matrix operand order repaired and checked by reading, not compiled, not run on hardware
//
#include <hip/hip_runtime.h>
#include <hip/hip_bf16.h>

typedef __attribute__((ext_vector_type(16))) __bf16 v16bf;
typedef __attribute__((ext_vector_type(8)))  __bf16 v8bf;
typedef __attribute__((ext_vector_type(8)))  float  v8f;

#define T_STEPS 100
#define BATCH   256
#define K0      784
#define K0P     800      // pad 784 -> multiple of 32
#define N0      1024
#define N1      1024
#define N2      10
#define N2P     16       // pad 10 -> 16 for WMMA tile

#define KT0     (K0P / 32)   // 25 k-tiles for layer 0
#define KT1     (N0 / 32)    // 32 k-tiles for layers 1,2
#define NT0     (N0 / 16)    // 64 n-tiles
#define NT1     (N1 / 16)    // 64

// LIF: v_dec = v + dt*tau*(v_leak - v + i) = 0.95*v + 0.05*i ; z = (v_dec-1)>0 ; v -= z
#define LIF_DECAY 0.95f
#define LIF_GAIN  0.05f

// ---------------------------------------------------------------------------
// A fragment 16x32 bf16 (wave32; this code's contiguous k order, same as packed B):
// lanes 0-15 = rows M=0..15, K {0..15}; lanes 16-31 same rows, K {16..31}
// Two 128-bit contiguous loads per lane.
// ---------------------------------------------------------------------------
__device__ inline v16bf load_a_frag(const __bf16* A, int lda, int row0, int k0) {
    int lane = threadIdx.x & 31;
    int row  = row0 + (lane & 15);
    int klo  = k0 + ((lane & 16) ? 16 : 0);
    int khi  = k0 + ((lane & 16) ? 24 : 8);
    const __bf16* base = A + (long)row * lda;
    v8bf lo = *(const v8bf*)(base + klo);
    v8bf hi = *(const v8bf*)(base + khi);
    v16bf a;
#pragma unroll
    for (int e = 0; e < 8; ++e) { a[e] = lo[e]; a[8 + e] = hi[e]; }
    return a;
}

// B fragment from pre-packed weights: Wp[((nt*KT + kt)*32 + lane)*16 + e]
// -> one 32-byte contiguous vector load per lane (2 x global_load_b128).
__device__ inline v16bf load_b_packed(const __bf16* Wp, int kt, int nt, int KT) {
    int lane = threadIdx.x & 31;
    return *(const v16bf*)(Wp + (((long)nt * KT + kt) * 32 + lane) * 16);
}

// ---------------------------------------------------------------------------
// Prep kernels
// ---------------------------------------------------------------------------
__global__ void pad_x_kernel(const float* __restrict__ x, __bf16* __restrict__ xp, long total) {
    long i = (long)blockIdx.x * blockDim.x + threadIdx.x;
    if (i >= total) return;
    int  k   = (int)(i % K0P);
    long row = i / K0P;
    float v = (k < K0) ? x[row * K0 + k] : 0.0f;
    xp[i] = (__bf16)v;
}

// Pack W [N,K] row-major fp32 into WMMA B-fragment layout:
// element e of lane L in (nt,kt) block = W[nt*16 + (L&15)][kt*32 + ((L&16)?16:0) + e]
__global__ void pack_w_kernel(const float* __restrict__ W, __bf16* __restrict__ Wp,
                              int N, int K, int NT, int KT) {
    long total = (long)NT * KT * 32 * 16;
    long i = (long)blockIdx.x * blockDim.x + threadIdx.x;
    if (i >= total) return;
    int e    = (int)(i & 15);
    int lane = (int)((i >> 4) & 31);
    int kt   = (int)((i >> 9) % KT);
    int nt   = (int)((i >> 9) / KT);
    int col  = nt * 16 + (lane & 15);
    int kk   = kt * 32 + ((lane & 16) ? 16 : 0) + e;
    float v = (col < N && kk < K) ? W[(long)col * K + kk] : 0.0f;
    Wp[i] = (__bf16)v;
}

__global__ void zero_f32_kernel(float* __restrict__ p, long n) {
    long i = (long)blockIdx.x * blockDim.x + threadIdx.x;
    if (i < n) p[i] = 0.0f;
}

// ---------------------------------------------------------------------------
// GEMM0 (batched over all T): I0[25600,1024] = Xpad @ W0^T + b0
// One wave per 16x64 strip: A fragment reused across 4 B fragments.
// ---------------------------------------------------------------------------
__global__ void __launch_bounds__(256)
gemm0_kernel(const __bf16* __restrict__ Xp, const __bf16* __restrict__ W0p,
             const float* __restrict__ b0, float* __restrict__ I0) {
    int wave = threadIdx.x >> 5;
    int gs   = blockIdx.x * 8 + wave;           // strip index
    const int NSTRIPS = N0 / 64;                // 16
    int m0  = (gs / NSTRIPS) * 16;
    int nt0 = (gs % NSTRIPS) * 4;
    v8f acc0 = {}, acc1 = {}, acc2 = {}, acc3 = {};
    for (int kt = 0; kt < KT0; ++kt) {
        v16bf a  = load_a_frag(Xp, K0P, m0, kt * 32);
        v16bf bb0 = load_b_packed(W0p, kt, nt0 + 0, KT0);
        v16bf bb1 = load_b_packed(W0p, kt, nt0 + 1, KT0);
        v16bf bb2 = load_b_packed(W0p, kt, nt0 + 2, KT0);
        v16bf bb3 = load_b_packed(W0p, kt, nt0 + 3, KT0);
        acc0 = __builtin_amdgcn_wmma_f32_16x16x32_bf16(false, a, false, bb0, (short)0, acc0, false, false);
        acc1 = __builtin_amdgcn_wmma_f32_16x16x32_bf16(false, a, false, bb1, (short)0, acc1, false, false);
        acc2 = __builtin_amdgcn_wmma_f32_16x16x32_bf16(false, a, false, bb2, (short)0, acc2, false, false);
        acc3 = __builtin_amdgcn_wmma_f32_16x16x32_bf16(false, a, false, bb3, (short)0, acc3, false, false);
    }
    int lane = threadIdx.x & 31;
    int r0   = m0 + ((lane & 16) ? 8 : 0);
    v8f accs[4] = {acc0, acc1, acc2, acc3};
#pragma unroll
    for (int j = 0; j < 4; ++j) {
        int col = (nt0 + j) * 16 + (lane & 15);
        float bias = b0[col];
#pragma unroll
        for (int r = 0; r < 8; ++r)
            I0[(long)(r0 + r) * N0 + col] = accs[j][r] + bias;
    }
}

// ---------------------------------------------------------------------------
// LIF for layer 0 at one timestep (elementwise over B*N0)
// ---------------------------------------------------------------------------
__global__ void lif0_kernel(const float* __restrict__ I0t, float* __restrict__ v0,
                            __bf16* __restrict__ S0, float* __restrict__ sp_out) {
    int i = blockIdx.x * blockDim.x + threadIdx.x;   // < BATCH*N0
    float v    = v0[i];
    float cur  = I0t[i];
    float vdec = LIF_DECAY * v + LIF_GAIN * cur;
    float z    = (vdec - 1.0f) > 0.0f ? 1.0f : 0.0f;
    v0[i]     = vdec - z;
    S0[i]     = (__bf16)z;
    sp_out[i] = z;
}

// ---------------------------------------------------------------------------
// GEMM1 + fused LIF1: I1 = S0[256,1024] @ W1^T + b1, LIF on v1, 16x64 strips
// ---------------------------------------------------------------------------
__global__ void __launch_bounds__(256)
gemm1_lif1_kernel(const __bf16* __restrict__ S0, const __bf16* __restrict__ W1p,
                  const float* __restrict__ b1, float* __restrict__ v1,
                  __bf16* __restrict__ S1, float* __restrict__ sp_out) {
    int wave = threadIdx.x >> 5;
    int gs   = blockIdx.x * 8 + wave;
    const int NSTRIPS = N1 / 64;                // 16
    int m0  = (gs / NSTRIPS) * 16;
    int nt0 = (gs % NSTRIPS) * 4;
    v8f acc0 = {}, acc1 = {}, acc2 = {}, acc3 = {};
    for (int kt = 0; kt < KT1; ++kt) {
        v16bf a  = load_a_frag(S0, N0, m0, kt * 32);
        v16bf bb0 = load_b_packed(W1p, kt, nt0 + 0, KT1);
        v16bf bb1 = load_b_packed(W1p, kt, nt0 + 1, KT1);
        v16bf bb2 = load_b_packed(W1p, kt, nt0 + 2, KT1);
        v16bf bb3 = load_b_packed(W1p, kt, nt0 + 3, KT1);
        acc0 = __builtin_amdgcn_wmma_f32_16x16x32_bf16(false, a, false, bb0, (short)0, acc0, false, false);
        acc1 = __builtin_amdgcn_wmma_f32_16x16x32_bf16(false, a, false, bb1, (short)0, acc1, false, false);
        acc2 = __builtin_amdgcn_wmma_f32_16x16x32_bf16(false, a, false, bb2, (short)0, acc2, false, false);
        acc3 = __builtin_amdgcn_wmma_f32_16x16x32_bf16(false, a, false, bb3, (short)0, acc3, false, false);
    }
    int lane = threadIdx.x & 31;
    int r0   = m0 + ((lane & 16) ? 8 : 0);
    v8f accs[4] = {acc0, acc1, acc2, acc3};
#pragma unroll
    for (int j = 0; j < 4; ++j) {
        int col = (nt0 + j) * 16 + (lane & 15);
        float bias = b1[col];
#pragma unroll
        for (int r = 0; r < 8; ++r) {
            long  idx  = (long)(r0 + r) * N1 + col;
            float cur  = accs[j][r] + bias;
            float v    = v1[idx];
            float vdec = LIF_DECAY * v + LIF_GAIN * cur;
            float z    = (vdec - 1.0f) > 0.0f ? 1.0f : 0.0f;
            v1[idx]     = vdec - z;
            S1[idx]     = (__bf16)z;
            sp_out[idx] = z;
        }
    }
}

// ---------------------------------------------------------------------------
// GEMM2 + fused LIF2: I2 = S1[256,1024] @ W2^T (N padded to 16), cols 0..9 valid
// ---------------------------------------------------------------------------
__global__ void __launch_bounds__(256)
gemm2_lif2_kernel(const __bf16* __restrict__ S1, const __bf16* __restrict__ W2p,
                  const float* __restrict__ b2, float* __restrict__ v2,
                  float* __restrict__ out_t, float* __restrict__ sp2_out) {
    int wave  = threadIdx.x >> 5;
    int gtile = blockIdx.x * 8 + wave;          // 0..15 row tiles, single col tile
    int m0 = gtile * 16;
    v8f acc = {};
    for (int kt = 0; kt < KT1; ++kt) {
        v16bf a = load_a_frag(S1, N1, m0, kt * 32);
        v16bf b = load_b_packed(W2p, kt, 0, KT1);
        acc = __builtin_amdgcn_wmma_f32_16x16x32_bf16(false, a, false, b, (short)0, acc, false, false);
    }
    int lane = threadIdx.x & 31;
    int col  = lane & 15;
    int r0   = m0 + ((lane & 16) ? 8 : 0);
    if (col < N2) {
        float bias = b2[col];
#pragma unroll
        for (int r = 0; r < 8; ++r) {
            int   row  = r0 + r;
            int   vidx = row * N2P + col;
            float cur  = acc[r] + bias;
            float v    = v2[vidx];
            float vdec = LIF_DECAY * v + LIF_GAIN * cur;
            float z    = (vdec - 1.0f) > 0.0f ? 1.0f : 0.0f;
            v2[vidx]                = vdec - z;
            out_t[row * N2 + col]   = z;
            sp2_out[row * N2 + col] = z;
        }
    }
}

// ---------------------------------------------------------------------------
extern "C" void kernel_launch(void* const* d_in, const int* in_sizes, int n_in,
                              void* d_out, int out_size, void* d_ws, size_t ws_size,
                              hipStream_t stream) {
    const float* inp = (const float*)d_in[0];
    const float* W0  = (const float*)d_in[1];
    const float* b0  = (const float*)d_in[2];
    const float* W1  = (const float*)d_in[3];
    const float* b1  = (const float*)d_in[4];
    const float* W2  = (const float*)d_in[5];
    const float* b2  = (const float*)d_in[6];
    float* out = (float*)d_out;

    // ---- workspace carve-out (256B aligned) ----
    char* ws = (char*)d_ws;
    size_t off = 0;
    auto carve = [&](size_t bytes) -> char* {
        char* p = ws + off;
        off = (off + bytes + 255) & ~(size_t)255;
        return p;
    };
    __bf16* Xpad = (__bf16*)carve((size_t)T_STEPS * BATCH * K0P * 2);
    __bf16* W0p  = (__bf16*)carve((size_t)NT0 * KT0 * 32 * 16 * 2);
    __bf16* W1p  = (__bf16*)carve((size_t)NT1 * KT1 * 32 * 16 * 2);
    __bf16* W2p  = (__bf16*)carve((size_t)1   * KT1 * 32 * 16 * 2);
    float*  I0   = (float*)carve((size_t)T_STEPS * BATCH * N0 * 4);
    float*  vblk = (float*)carve((size_t)BATCH * (N0 + N1 + N2P) * 4);
    float*  v0 = vblk;
    float*  v1 = vblk + (size_t)BATCH * N0;
    float*  v2 = vblk + (size_t)BATCH * (N0 + N1);
    __bf16* S0 = (__bf16*)carve((size_t)BATCH * N0 * 2);
    __bf16* S1 = (__bf16*)carve((size_t)BATCH * N1 * 2);

    // ---- d_out layout ----
    const long OUT_ELEMS = (long)T_STEPS * BATCH * N2;              // 256,000
    const long STEP_SP   = (long)BATCH * (N0 + N1 + N2);            // 526,848
    float* spikes = out + OUT_ELEMS;

    // ---- prep ----
    {
        long totX = (long)T_STEPS * BATCH * K0P;
        pad_x_kernel<<<(int)((totX + 255) / 256), 256, 0, stream>>>(inp, Xpad, totX);
        long p0 = (long)NT0 * KT0 * 512;
        pack_w_kernel<<<(int)((p0 + 255) / 256), 256, 0, stream>>>(W0, W0p, N0, K0, NT0, KT0);
        long p1 = (long)NT1 * KT1 * 512;
        pack_w_kernel<<<(int)((p1 + 255) / 256), 256, 0, stream>>>(W1, W1p, N1, N0, NT1, KT1);
        long p2 = (long)1 * KT1 * 512;
        pack_w_kernel<<<(int)((p2 + 255) / 256), 256, 0, stream>>>(W2, W2p, N2, N1, 1, KT1);
        long nv = (long)BATCH * (N0 + N1 + N2P);
        zero_f32_kernel<<<(int)((nv + 255) / 256), 256, 0, stream>>>(vblk, nv);
    }

    // ---- batched layer-0 GEMM over all timesteps ----
    {
        int strips = (T_STEPS * BATCH / 16) * (N0 / 64);   // 1600*16 = 25600
        gemm0_kernel<<<strips / 8, 256, 0, stream>>>(Xpad, W0p, b0, I0);
    }

    // ---- sequential timestep loop (layers are fused GEMM+LIF) ----
    for (int t = 0; t < T_STEPS; ++t) {
        float* sp_base = spikes + (long)t * STEP_SP;
        const float* I0t = I0 + (long)t * BATCH * N0;

        lif0_kernel<<<(BATCH * N0) / 256, 256, 0, stream>>>(I0t, v0, S0, sp_base);

        int strips1 = (BATCH / 16) * (N1 / 64);            // 16*16 = 256
        gemm1_lif1_kernel<<<strips1 / 8, 256, 0, stream>>>(S0, W1p, b1, v1, S1,
                                                           sp_base + (long)BATCH * N0);

        gemm2_lif2_kernel<<<2, 256, 0, stream>>>(S1, W2p, b2, v2,
                                                 out + (long)t * BATCH * N2,
                                                 sp_base + (long)BATCH * (N0 + N1));
    }
}
